// gruEncoder_86449101734130
// MI455X (gfx1250) — matrix operand order repaired and checked by reading, not compiled, not run on hardware
//
#include <hip/hip_runtime.h>
#include <cstdint>
#include <cstddef>

// ---------------- CDNA5 vector types ----------------
typedef __bf16 v16bf __attribute__((ext_vector_type(16)));
typedef __bf16 v8bf  __attribute__((ext_vector_type(8)));
typedef float  v8f   __attribute__((ext_vector_type(8)));
typedef int    v4i   __attribute__((ext_vector_type(4)));

// ---------------- async global->LDS copy (gfx1250) ----------------
#if defined(__has_builtin)
#  if __has_builtin(__builtin_amdgcn_global_load_async_to_lds_b128) && \
      __has_builtin(__builtin_amdgcn_s_wait_asynccnt)
#    define USE_ASYNC_COPY 1
#  endif
#endif
#ifndef USE_ASYNC_COPY
#  define USE_ASYNC_COPY 0
#endif

__device__ __forceinline__ void cp16_to_lds(const __bf16* gsrc, __bf16* ldst) {
#if USE_ASYNC_COPY
  __builtin_amdgcn_global_load_async_to_lds_b128(
      (__attribute__((address_space(1))) v4i*)gsrc,
      (__attribute__((address_space(3))) v4i*)ldst, 0, 0);
#else
  *(uint4*)ldst = *(const uint4*)gsrc;
#endif
}

__device__ __forceinline__ void wait_async_copies() {
#if USE_ASYNC_COPY
  __builtin_amdgcn_s_wait_asynccnt(0);
#endif
}

// ---------------- WMMA GEMM: C[M,N] = A[M,K] (bf16) * W[N,K]^T (bf16) + bias[N] ----------------
// Block tile 128x128, KT=32, 256 threads = 8 wave32 waves, each wave -> 32x64 (2x4 wmma tiles).
#define MT 128
#define NT 128
#define KT 32
#define LDS_STRIDE 40   // KT + 8 bf16 pad -> 80B rows, 16B aligned

__global__ __launch_bounds__(256) void gemm_bf16_wmma(
    const __bf16* __restrict__ A, const __bf16* __restrict__ W,
    const float* __restrict__ bias, float* __restrict__ C,
    int M, int N, int K) {
  __shared__ __align__(16) __bf16 ldsA[2][MT * LDS_STRIDE];
  __shared__ __align__(16) __bf16 ldsB[2][NT * LDS_STRIDE];

  const int tid  = (int)threadIdx.x;
  const int lane = tid & 31;
  const int wave = tid >> 5;   // 0..7
  const int wm   = wave >> 1;  // 0..3 : 32-row strip
  const int wn   = wave & 1;   // 0..1 : 64-col strip
  const int l16  = lane & 15;
  const int kh   = lane >> 4;  // K-half select

  const int n0 = (int)blockIdx.x * NT;
  const int m0 = (int)blockIdx.y * MT;

  const int ldRow = tid >> 1;          // 0..127
  const int ldSeg = (tid & 1) << 4;    // 0 or 16 (bf16 units)

  const __bf16* gA = A + (size_t)(m0 + ldRow) * K + ldSeg;
  const __bf16* gB = W + (size_t)(n0 + ldRow) * K + ldSeg;

  auto load_tile = [&](int buf, int k0) {
    const __bf16* ga = gA + k0;
    const __bf16* gb = gB + k0;
    __bf16* la = &ldsA[buf][ldRow * LDS_STRIDE + ldSeg];
    __bf16* lb = &ldsB[buf][ldRow * LDS_STRIDE + ldSeg];
    cp16_to_lds(ga,     la);
    cp16_to_lds(ga + 8, la + 8);
    cp16_to_lds(gb,     lb);
    cp16_to_lds(gb + 8, lb + 8);
  };

  v8f acc[2][4];
#pragma unroll
  for (int i = 0; i < 2; ++i)
#pragma unroll
    for (int j = 0; j < 4; ++j)
#pragma unroll
      for (int q = 0; q < 8; ++q) acc[i][j][q] = 0.0f;

  const int nk = K >> 5;
  load_tile(0, 0);

  for (int kt = 0; kt < nk; ++kt) {
    wait_async_copies();
    __syncthreads();
    if (kt + 1 < nk) load_tile((kt + 1) & 1, (kt + 1) << 5);

    const int buf = kt & 1;
    union Frag { v16bf v; v8bf h[2]; };
    Frag afrag[2], bfrag;

#pragma unroll
    for (int ms = 0; ms < 2; ++ms) {
      // A 16x32 layout: lanes 0-15 -> K {0..7,16..23}; lanes 16-31 -> K {8..15,24..31}
      const __bf16* ap = &ldsA[buf][(wm * 32 + ms * 16 + l16) * LDS_STRIDE + (kh << 3)];
      afrag[ms].h[0] = *(const v8bf*)(ap);
      afrag[ms].h[1] = *(const v8bf*)(ap + 16);
    }

#pragma unroll
    for (int ns = 0; ns < 4; ++ns) {
      // B 32x16 layout: lanes 0-15 -> K {0..7,16..23}; lanes 16-31 -> K {8..15,24..31}
      const __bf16* bp = &ldsB[buf][(wn * 64 + ns * 16 + l16) * LDS_STRIDE + (kh << 3)];
      bfrag.h[0] = *(const v8bf*)(bp);
      bfrag.h[1] = *(const v8bf*)(bp + 16);
#pragma unroll
      for (int ms = 0; ms < 2; ++ms) {
        acc[ms][ns] = __builtin_amdgcn_wmma_f32_16x16x32_bf16(
            false, afrag[ms].v, false, bfrag.v, (short)0, acc[ms][ns], false, false);
      }
    }
  }

  // Epilogue: C layout -> m = v + 8*(lane>=16), n = lane&15
  const int mW = m0 + wm * 32;
  const int nW = n0 + wn * 64;
#pragma unroll
  for (int ms = 0; ms < 2; ++ms) {
#pragma unroll
    for (int ns = 0; ns < 4; ++ns) {
      const int col = nW + ns * 16 + l16;
      const float bv = bias ? bias[col] : 0.0f;
      const int rbase = mW + ms * 16 + (kh << 3);
      float* crow = C + (size_t)rbase * N + col;
#pragma unroll
      for (int v = 0; v < 8; ++v) crow[(size_t)v * N] = acc[ms][ns][v] + bv;
    }
  }
}

// ---------------- elementwise / reduction kernels ----------------
__global__ void cvt_f32_to_bf16(const float* __restrict__ src, __bf16* __restrict__ dst, size_t n) {
  size_t i = (size_t)blockIdx.x * blockDim.x + threadIdx.x;
  const size_t stride = (size_t)gridDim.x * blockDim.x;
  for (; i < n; i += stride) dst[i] = (__bf16)src[i];
}

__global__ void gru_pointwise(const float* __restrict__ gi, const float* __restrict__ gh,
                              const float* __restrict__ hprev, float* __restrict__ nh,
                              __bf16* __restrict__ nhb, int Bn, int Hn) {
  const size_t n = (size_t)Bn * Hn;
  size_t i = (size_t)blockIdx.x * blockDim.x + threadIdx.x;
  const size_t stride = (size_t)gridDim.x * blockDim.x;
  for (; i < n; i += stride) {
    const size_t b = i / Hn, j = i % Hn;
    const size_t base = b * (size_t)(3 * Hn);
    const float ir = gi[base + j],           hr = gh[base + j];
    const float iz = gi[base + Hn + j],      hz = gh[base + Hn + j];
    const float in_ = gi[base + 2 * Hn + j], hn = gh[base + 2 * Hn + j];
    const float r = 1.0f / (1.0f + __expf(-(ir + hr)));
    const float z = 1.0f / (1.0f + __expf(-(iz + hz)));
    const float nv = tanhf(in_ + r * hn);
    const float out = (1.0f - z) * nv + z * hprev[i];
    nh[i] = out;
    if (nhb) nhb[i] = (__bf16)out;
  }
}

// actKind: 0 = ELU then LN (optionally affine), 1 = Mish then LN
__global__ __launch_bounds__(256) void act_ln(
    const float* __restrict__ x, const float* __restrict__ gamma,
    const float* __restrict__ beta, float* __restrict__ of,
    __bf16* __restrict__ ob, int Hn, int actKind) {
  const int row = blockIdx.x, t = threadIdx.x;
  const float* xr = x + (size_t)row * Hn;
  float v[8];
  int cnt = 0;
  float s = 0.0f, s2 = 0.0f;
  for (int c = t; c < Hn; c += 256) {
    float a = xr[c];
    if (actKind == 0) {
      a = (a > 0.0f) ? a : expm1f(a);
    } else {
      const float sp = (a > 20.0f) ? a : log1pf(__expf(a));
      a = a * tanhf(sp);
    }
    v[cnt++] = a; s += a; s2 += a * a;
  }
  __shared__ float r1[256], r2[256];
  r1[t] = s; r2[t] = s2;
  __syncthreads();
  for (int off = 128; off > 0; off >>= 1) {
    if (t < off) { r1[t] += r1[t + off]; r2[t] += r2[t + off]; }
    __syncthreads();
  }
  const float mean = r1[0] / Hn;
  const float var  = r2[0] / Hn - mean * mean;
  const float inv  = rsqrtf(var + 1e-5f);
  cnt = 0;
  for (int c = t; c < Hn; c += 256) {
    float o = (v[cnt++] - mean) * inv;
    if (gamma) o = o * gamma[c] + beta[c];
    if (of) of[(size_t)row * Hn + c] = o;
    if (ob) ob[(size_t)row * Hn + c] = (__bf16)o;
  }
}

__global__ void mish_bf16(const float* __restrict__ x, __bf16* __restrict__ ob, size_t n) {
  size_t i = (size_t)blockIdx.x * blockDim.x + threadIdx.x;
  const size_t stride = (size_t)gridDim.x * blockDim.x;
  for (; i < n; i += stride) {
    const float a = x[i];
    const float sp = (a > 20.0f) ? a : log1pf(__expf(a));
    ob[i] = (__bf16)(a * tanhf(sp));
  }
}

__global__ void gate_softmax(const float* __restrict__ y, const float* __restrict__ Wg,
                             const float* __restrict__ bg, float* __restrict__ gate, int Hn) {
  const int b = blockIdx.x;
  const int t = threadIdx.x;   // 64 threads
  const int e = t >> 3, s = t & 7;
  const float* yr = y + (size_t)b * Hn;
  const float* wr = Wg + (size_t)e * Hn;
  float p = 0.0f;
  for (int k = s; k < Hn; k += 8) p += yr[k] * wr[k];
  __shared__ float red[64];
  red[t] = p;
  __syncthreads();
  if (s == 0) {
    float vsum = 0.0f;
    for (int i = 0; i < 8; ++i) vsum += red[e * 8 + i];
    red[e * 8] = vsum + bg[e];
  }
  __syncthreads();
  if (t == 0) {
    float vals[8]; float mx = -1e30f;
    for (int q = 0; q < 8; ++q) { vals[q] = red[q * 8]; mx = fmaxf(mx, vals[q]); }
    float sum = 0.0f;
    for (int q = 0; q < 8; ++q) { vals[q] = __expf(vals[q] - mx); sum += vals[q]; }
    const float inv = 1.0f / sum;
    for (int q = 0; q < 8; ++q) gate[(size_t)b * 8 + q] = vals[q] * inv;
  }
}

__global__ void combine_expert(const float* __restrict__ o, const float* __restrict__ gate,
                               float* __restrict__ z, int e, int En, int Ln, size_t n,
                               int overwrite) {
  size_t i = (size_t)blockIdx.x * blockDim.x + threadIdx.x;
  const size_t stride = (size_t)gridDim.x * blockDim.x;
  for (; i < n; i += stride) {
    const size_t b = i / Ln;
    const float val = gate[b * En + e] * o[i];
    z[i] = overwrite ? val : (z[i] + val);
  }
}

// ---------------- host driver ----------------
extern "C" void kernel_launch(void* const* d_in, const int* in_sizes, int n_in,
                              void* d_out, int out_size, void* d_ws, size_t ws_size,
                              hipStream_t stream) {
  constexpr int B = 8192, I = 1024, H = 1024, L = 256, E = 8;
  constexpr int H3 = 3 * H;
  const size_t BH = (size_t)B * H;

  const float* x    = (const float*)d_in[0];
  const float* hid  = (const float*)d_in[1];
  const float* Wih0 = (const float*)d_in[2];
  const float* Whh0 = (const float*)d_in[3];
  const float* bih0 = (const float*)d_in[4];
  const float* bhh0 = (const float*)d_in[5];
  const float* Wih1 = (const float*)d_in[6];
  const float* Whh1 = (const float*)d_in[7];
  const float* bih1 = (const float*)d_in[8];
  const float* bhh1 = (const float*)d_in[9];
  const float* ln1g = (const float*)d_in[10];
  const float* ln1b = (const float*)d_in[11];
  const float* Wg   = (const float*)d_in[12];
  const float* bg   = (const float*)d_in[13];
  const float* W1   = (const float*)d_in[14];
  const float* b1   = (const float*)d_in[15];
  const float* W2   = (const float*)d_in[16];
  const float* b2   = (const float*)d_in[17];
  const float* W3   = (const float*)d_in[18];
  const float* b3   = (const float*)d_in[19];
  const float* W4   = (const float*)d_in[20];
  const float* b4   = (const float*)d_in[21];

  float* out_z   = (float*)d_out;             // [B, L]
  float* out_nh0 = out_z + (size_t)B * L;     // [B, H]
  float* out_nh1 = out_nh0 + BH;              // [B, H]

  char* ws = (char*)d_ws;
  size_t off = 0;
  auto carve = [&](size_t bytes) -> void* {
    void* p = ws + off;
    off += (bytes + 255) & ~(size_t)255;
    return p;
  };

  __bf16* xb    = (__bf16*)carve((size_t)B * I * 2);
  __bf16* h0b   = (__bf16*)carve(BH * 2);
  __bf16* h1b   = (__bf16*)carve(BH * 2);
  __bf16* nh0b  = (__bf16*)carve(BH * 2);
  __bf16* yb    = (__bf16*)carve(BH * 2);
  __bf16* t1b   = (__bf16*)carve(BH * 2);
  __bf16* t2b   = (__bf16*)carve(BH * 2);
  __bf16* Wih0b = (__bf16*)carve((size_t)H3 * I * 2);
  __bf16* Whh0b = (__bf16*)carve((size_t)H3 * H * 2);
  __bf16* Wih1b = (__bf16*)carve((size_t)H3 * H * 2);
  __bf16* Whh1b = (__bf16*)carve((size_t)H3 * H * 2);
  __bf16* W1b   = (__bf16*)carve((size_t)E * H * H * 2);
  __bf16* W2b   = (__bf16*)carve((size_t)E * H * H * 2);
  __bf16* W3b   = (__bf16*)carve((size_t)E * H * H * 2);
  __bf16* W4b   = (__bf16*)carve((size_t)E * L * H * 2);
  float* gi   = (float*)carve((size_t)B * H3 * 4);
  float* gh   = (float*)carve((size_t)B * H3 * 4);
  float* yf   = (float*)carve(BH * 4);
  float* gate = (float*)carve((size_t)B * E * 4);
  float* t1f  = (float*)carve(BH * 4);
  float* t2f  = (float*)carve(BH * 4);
  float* of   = (float*)carve((size_t)B * L * 4);

  auto cvt = [&](const float* s, __bf16* d, size_t n) {
    cvt_f32_to_bf16<<<dim3(2048), dim3(256), 0, stream>>>(s, d, n);
  };
  cvt(x, xb, (size_t)B * I);
  cvt(hid, h0b, BH);
  cvt(hid + BH, h1b, BH);
  cvt(Wih0, Wih0b, (size_t)H3 * I);
  cvt(Whh0, Whh0b, (size_t)H3 * H);
  cvt(Wih1, Wih1b, (size_t)H3 * H);
  cvt(Whh1, Whh1b, (size_t)H3 * H);
  cvt(W1, W1b, (size_t)E * H * H);
  cvt(W2, W2b, (size_t)E * H * H);
  cvt(W3, W3b, (size_t)E * H * H);
  cvt(W4, W4b, (size_t)E * L * H);

  auto gemm = [&](const __bf16* Aq, const __bf16* Wq, const float* bq, float* Cq,
                  int M, int N, int K) {
    dim3 g(N / 128, M / 128);
    gemm_bf16_wmma<<<g, dim3(256), 0, stream>>>(Aq, Wq, bq, Cq, M, N, K);
  };

  // ---- GRU layer 0 ----
  gemm(xb, Wih0b, bih0, gi, B, H3, I);
  gemm(h0b, Whh0b, bhh0, gh, B, H3, H);
  gru_pointwise<<<dim3(8192), dim3(256), 0, stream>>>(gi, gh, hid, out_nh0, nh0b, B, H);

  // ---- GRU layer 1 ----
  gemm(nh0b, Wih1b, bih1, gi, B, H3, H);
  gemm(h1b, Whh1b, bhh1, gh, B, H3, H);
  gru_pointwise<<<dim3(8192), dim3(256), 0, stream>>>(gi, gh, hid + BH, out_nh1,
                                                      (__bf16*)nullptr, B, H);

  // ---- y = LN(elu(nh1)) * g + b ----
  act_ln<<<dim3(B), dim3(256), 0, stream>>>(out_nh1, ln1g, ln1b, yf, yb, H, 0);

  // ---- gating ----
  gate_softmax<<<dim3(B), dim3(64), 0, stream>>>(yf, Wg, bg, gate, H);

  // ---- experts (dense over E, gated accumulate) ----
  for (int e = 0; e < E; ++e) {
    gemm(yb, W1b + (size_t)e * H * H, b1 + (size_t)e * H, t1f, B, H, H);
    act_ln<<<dim3(B), dim3(256), 0, stream>>>(t1f, (const float*)nullptr,
                                              (const float*)nullptr, (float*)nullptr,
                                              t1b, H, 1);
    gemm(t1b, W2b + (size_t)e * H * H, b2 + (size_t)e * H, t2f, B, H, H);
    act_ln<<<dim3(B), dim3(256), 0, stream>>>(t2f, (const float*)nullptr,
                                              (const float*)nullptr, (float*)nullptr,
                                              t2b, H, 1);
    gemm(t2b, W3b + (size_t)e * H * H, b3 + (size_t)e * H, t1f, B, H, H);
    mish_bf16<<<dim3(4096), dim3(256), 0, stream>>>(t1f, t1b, BH);
    gemm(t1b, W4b + (size_t)e * L * H, b4 + (size_t)e * L, of, B, L, H);
    combine_expert<<<dim3(2048), dim3(256), 0, stream>>>(of, gate, out_z, e, E, L,
                                                         (size_t)B * L, e == 0 ? 1 : 0);
  }
}
